// Attention_39711267619462
// MI455X (gfx1250) — matrix operand order repaired and checked by reading, not compiled, not run on hardware
//
#include <hip/hip_runtime.h>

typedef __attribute__((ext_vector_type(16))) _Float16 v16h;
typedef __attribute__((ext_vector_type(8)))  float    v8f;

static constexpr int Bb = 4, Nn = 2048, DIM = 128, HEADS = 8, HEAD = 128, INNER = 1024;
static constexpr int NQKV = 3 * INNER * DIM;     // 393216 w_qkv elements
static constexpr int NWO  = DIM * INNER;         // 131072 w_out elements

union Frag {
    v16h v;
    _Float16 h[16];
    unsigned int u[8];
    uint4 q[2];
};

__device__ __forceinline__ v8f wmma_f16(const Frag& a, const Frag& b, v8f c) {
    return __builtin_amdgcn_wmma_f32_16x16x32_f16(false, a.v, false, b.v,
                                                  (short)0, c, false, false);
}

// ---------------------------------------------------------------------------
// Kernel 0a: LayerNorm once into a global f16 activation buffer xn[8192][128].
//   grid = 256 blocks x 256 threads; 8 threads per row, shuffle reductions.
// ---------------------------------------------------------------------------
__global__ __launch_bounds__(256) void k_ln(
    const float* __restrict__ x, const float* __restrict__ gam,
    const float* __restrict__ bet, _Float16* __restrict__ xnw)
{
    const int tid = threadIdx.x;
    const int row = tid >> 3, sub = tid & 7;
    const long R = (long)blockIdx.x * 32 + row;
    const float* xr = x + R * DIM + sub * 16;
    float xv[16], s = 0.f, s2 = 0.f;
    #pragma unroll
    for (int i = 0; i < 16; ++i) { float v = xr[i]; xv[i] = v; s += v; s2 += v * v; }
    #pragma unroll
    for (int msk = 1; msk < 8; msk <<= 1) { s += __shfl_xor(s, msk); s2 += __shfl_xor(s2, msk); }
    const float mean = s * (1.f / DIM);
    const float rstd = rsqrtf(s2 * (1.f / DIM) - mean * mean + 1e-5f);
    #pragma unroll
    for (int i = 0; i < 16; ++i) {
        const int e = sub * 16 + i;
        xnw[R * DIM + e] = (_Float16)((xv[i] - mean) * rstd * gam[e] + bet[e]);
    }
}

// ---------------------------------------------------------------------------
// Kernel 0b: one-time fp32 -> f16 conversion of w_qkv and w_out.
//   grid = 512 blocks x 256 threads, 4 elements per thread (float4 in, 2 cvt_pk).
// ---------------------------------------------------------------------------
__global__ __launch_bounds__(256) void k_cvt(
    const float* __restrict__ wqkv, const float* __restrict__ wout,
    _Float16* __restrict__ wq16, _Float16* __restrict__ wo16)
{
    const long i4 = ((long)blockIdx.x * 256 + threadIdx.x) * 4;
    const float* src;
    _Float16* dst;
    long off;
    if (i4 < NQKV) { src = wqkv; dst = wq16; off = i4; }
    else           { src = wout; dst = wo16; off = i4 - NQKV; }
    const float4 f = *(const float4*)(src + off);
    dst[off + 0] = (_Float16)f.x;
    dst[off + 1] = (_Float16)f.y;
    dst[off + 2] = (_Float16)f.z;
    dst[off + 3] = (_Float16)f.w;
}

// ---------------------------------------------------------------------------
// Kernel 1: QKV GEMM (f16 WMMA, all-f16 operands) + RoPE epilogue.
//   grid = 3072 (256 row-blocks x 12 col-blocks), 256 threads (8 waves).
//   Wave: 16 rows x 64 cols, K=128 (4 wmma k-steps). Q scaled by HEAD^-0.5.
//   V stored transposed [b,h,d,n] for the PV GEMM.
// ---------------------------------------------------------------------------
__global__ __launch_bounds__(256) void k_qkv(
    const _Float16* __restrict__ xnw, const _Float16* __restrict__ wq16,
    _Float16* __restrict__ qw, _Float16* __restrict__ kw, _Float16* __restrict__ vw)
{
    const int tid = threadIdx.x;
    const int rowblk = blockIdx.x / 12;
    const int colblk = blockIdx.x % 12;
    const int wave = tid >> 5, lane = tid & 31;
    const int r = lane & 15, hh = lane >> 4;
    const long Rbase = (long)rowblk * 32 + (wave & 1) * 16;
    const int colbase = colblk * 256 + (wave >> 1) * 64;

    Frag A[4];
    #pragma unroll
    for (int ks = 0; ks < 4; ++ks) {               // A 16x32: K chunks at hh*8 and 16+hh*8
        const _Float16* p = xnw + (Rbase + r) * DIM + ks * 32 + hh * 8;
        A[ks].q[0] = *(const uint4*)p;
        A[ks].q[1] = *(const uint4*)(p + 16);
    }

    #pragma unroll
    for (int ct = 0; ct < 4; ++ct) {
        const int e0 = colbase + ct * 16 + r;      // this lane's B column (output channel)
        v8f acc = {};
        #pragma unroll
        for (int ks = 0; ks < 4; ++ks) {
            Frag Bf;                               // B 32x16: K chunks at hh*8 and 16+hh*8
            const _Float16* p = wq16 + (long)e0 * DIM + ks * 32 + hh * 8;
            Bf.q[0] = *(const uint4*)p;
            Bf.q[1] = *(const uint4*)(p + 16);
            acc = wmma_f16(A[ks], Bf, acc);
        }
        // ---- Epilogue: RoPE (+ Q scaling), scatter to q/k/v workspace ----
        const int sec = e0 >> 10;                  // 0=q 1=k 2=v
        const int eh = e0 & 1023;
        const int hhead = eh >> 7;
        const int d = eh & 127;
        const bool doRope = (sec < 2) && (hhead != HEADS - 1);
        const float invf = __expf(-(float)(d & ~1) * (9.210340371976184f / 128.f)); // 10000^(-2p/128)
        #pragma unroll
        for (int i = 0; i < 8; ++i) {
            const long R = Rbase + i + 8 * hh;
            const int b = (int)(R >> 11), n = (int)(R & 2047);
            float val = acc[i];
            const float partner = __shfl_xor(val, 1);   // RoPE pair lives in lane^1
            if (doRope) {
                float sn, cs; __sincosf((float)n * invf, &sn, &cs);
                val = (d & 1) ? (val * cs + partner * sn) : (val * cs - partner * sn);
            }
            if (sec == 0) {
                val *= 0.08838834764831845f;            // 1/sqrt(HEAD)
                qw[(((long)b * HEADS + hhead) * Nn + n) * HEAD + d] = (_Float16)val;
            } else if (sec == 1) {
                kw[(((long)b * HEADS + hhead) * Nn + n) * HEAD + d] = (_Float16)val;
            } else {
                vw[(((long)b * HEADS + hhead) * HEAD + d) * Nn + n] = (_Float16)val; // transposed
            }
        }
    }
}

// ---------------------------------------------------------------------------
// Kernel 2: flash attention per (b,h). grid = 1024 blocks, 128 threads (4 waves).
//   Wave owns 16 query rows, streams K/V in 32-column blocks with online softmax.
// ---------------------------------------------------------------------------
__global__ __launch_bounds__(128) void k_attn(
    const _Float16* __restrict__ qw, const _Float16* __restrict__ kw,
    const _Float16* __restrict__ vw, _Float16* __restrict__ ow)
{
    constexpr int PLD = 40;                        // P staging row stride (halves)
    __shared__ __align__(16) _Float16 pst[4 * 16 * PLD];

    const int tid = threadIdx.x;
    const int wave = tid >> 5, lane = tid & 31;
    const int r = lane & 15, hh = lane >> 4;
    const int bh = blockIdx.x >> 5;                // 0..31 = b*8+h
    const int qblk = blockIdx.x & 31;
    const int bat = bh >> 3, hd = bh & 7;
    const int q0 = qblk * 64 + wave * 16;

    const _Float16* qb = qw + (long)bh * Nn * HEAD;
    const _Float16* kb = kw + (long)bh * Nn * HEAD;
    const _Float16* vb = vw + (long)bh * HEAD * Nn;
    _Float16* pw = &pst[wave * 16 * PLD];

    Frag AQ[4];
    #pragma unroll
    for (int ks = 0; ks < 4; ++ks) {
        const _Float16* p = qb + (long)(q0 + r) * HEAD + ks * 32 + hh * 8;
        AQ[ks].q[0] = *(const uint4*)p;
        AQ[ks].q[1] = *(const uint4*)(p + 16);
    }

    float m[8], l[8];
    v8f O[8];
    const v8f vz = {};
    #pragma unroll
    for (int i = 0; i < 8; ++i) { m[i] = -1e30f; l[i] = 0.f; }
    #pragma unroll
    for (int nt = 0; nt < 8; ++nt) O[nt] = vz;

    for (int kv0 = 0; kv0 < Nn; kv0 += 32) {
        // speculative prefetch of the next K/V block (global_prefetch_b8)
        __builtin_prefetch(kb + (long)(kv0 + 32 + r) * HEAD, 0, 0);
        __builtin_prefetch(vb + (long)r * Nn + kv0 + 32, 0, 0);

        // ---- S = Q K^T (scale pre-folded into Q), two 16x16 tiles ----
        v8f S0 = vz, S1 = vz;
        #pragma unroll
        for (int ks = 0; ks < 4; ++ks) {
            Frag K0, K1;
            const _Float16* p0 = kb + (long)(kv0 + r) * HEAD + ks * 32 + hh * 8;
            K0.q[0] = *(const uint4*)p0;       K0.q[1] = *(const uint4*)(p0 + 16);
            const _Float16* p1 = p0 + 16 * HEAD;
            K1.q[0] = *(const uint4*)p1;       K1.q[1] = *(const uint4*)(p1 + 16);
            S0 = wmma_f16(AQ[ks], K0, S0);
            S1 = wmma_f16(AQ[ks], K1, S1);
        }
        // ---- online softmax: row max/sum across the 16-lane groups ----
        float alpha[8];
        #pragma unroll
        for (int i = 0; i < 8; ++i) {
            float mx = fmaxf(S0[i], S1[i]);
            #pragma unroll
            for (int msk = 1; msk < 16; msk <<= 1) mx = fmaxf(mx, __shfl_xor(mx, msk));
            const float mnew = fmaxf(m[i], mx);
            alpha[i] = __expf(m[i] - mnew);
            m[i] = mnew;
            const float p0 = __expf(S0[i] - mnew);
            const float p1 = __expf(S1[i] - mnew);
            const int row = i + 8 * hh;           // C-layout row for this VGPR
            pw[row * PLD + r]      = (_Float16)p0;
            pw[row * PLD + 16 + r] = (_Float16)p1;
            float ps = p0 + p1;
            #pragma unroll
            for (int msk = 1; msk < 16; msk <<= 1) ps += __shfl_xor(ps, msk);
            l[i] = l[i] * alpha[i] + ps;
        }
        asm volatile("s_wait_dscnt 0" ::: "memory");   // P staged in LDS before A-layout reads
        #pragma unroll
        for (int nt = 0; nt < 8; ++nt)
            #pragma unroll
            for (int i = 0; i < 8; ++i) O[nt][i] *= alpha[i];
        // ---- O += P V ----
        Frag AP;
        {
            const _Float16* p = &pw[r * PLD + hh * 8];
            AP.q[0] = *(const uint4*)p;
            AP.q[1] = *(const uint4*)(p + 16);
        }
        #pragma unroll
        for (int nt = 0; nt < 8; ++nt) {
            Frag VB;                               // V transposed: contiguous K run per column d
            const _Float16* p = vb + (long)(nt * 16 + r) * Nn + kv0 + hh * 8;
            VB.q[0] = *(const uint4*)p;  VB.q[1] = *(const uint4*)(p + 16);
            O[nt] = wmma_f16(AP, VB, O[nt]);
        }
    }
    // ---- normalize and store [B,N,INNER] f16 ----
    #pragma unroll
    for (int i = 0; i < 8; ++i) l[i] = 1.f / l[i];
    #pragma unroll
    for (int nt = 0; nt < 8; ++nt) {
        const int d = nt * 16 + r;
        #pragma unroll
        for (int i = 0; i < 8; ++i) {
            const int n = q0 + i + 8 * hh;
            ow[((long)bat * Nn + n) * INNER + hd * HEAD + d] = (_Float16)(O[nt][i] * l[i]);
        }
    }
}

// ---------------------------------------------------------------------------
// Kernel 3: out = o @ w_out^T + b_out. M=8192, Nout=128, K=1024 (32 k-steps).
//   grid = 128 blocks, 256 threads (8 waves); wave = 16 rows x 64 cols.
// ---------------------------------------------------------------------------
__global__ __launch_bounds__(256) void k_proj(
    const _Float16* __restrict__ ow, const _Float16* __restrict__ wo16,
    const float* __restrict__ bout, float* __restrict__ out)
{
    const int tid = threadIdx.x;
    const int wave = tid >> 5, lane = tid & 31;
    const int r = lane & 15, hh = lane >> 4;
    const int rowTile = wave >> 1;
    const int colbase = (wave & 1) * 64;
    const long Rbase = (long)blockIdx.x * 64 + rowTile * 16;

    v8f acc[4];
    const v8f vz = {};
    #pragma unroll
    for (int ct = 0; ct < 4; ++ct) acc[ct] = vz;

    for (int ks = 0; ks < 32; ++ks) {
        Frag A;
        const _Float16* p = ow + (Rbase + r) * INNER + ks * 32 + hh * 8;
        A.q[0] = *(const uint4*)p;
        A.q[1] = *(const uint4*)(p + 16);
        #pragma unroll
        for (int ct = 0; ct < 4; ++ct) {
            const int c = colbase + ct * 16 + r;
            Frag Bf;
            const _Float16* wp = wo16 + (long)c * INNER + ks * 32 + hh * 8;
            Bf.q[0] = *(const uint4*)wp;
            Bf.q[1] = *(const uint4*)(wp + 16);
            acc[ct] = wmma_f16(A, Bf, acc[ct]);
        }
    }
    #pragma unroll
    for (int ct = 0; ct < 4; ++ct) {
        const int c = colbase + ct * 16 + r;
        const float bo = bout[c];
        #pragma unroll
        for (int i = 0; i < 8; ++i) {
            const long R = Rbase + i + 8 * hh;
            out[R * DIM + c] = acc[ct][i] + bo;
        }
    }
}

// ---------------------------------------------------------------------------
extern "C" void kernel_launch(void* const* d_in, const int* in_sizes, int n_in,
                              void* d_out, int out_size, void* d_ws, size_t ws_size,
                              hipStream_t stream)
{
    const float* x    = (const float*)d_in[0];
    const float* gam  = (const float*)d_in[1];
    const float* bet  = (const float*)d_in[2];
    const float* wqkv = (const float*)d_in[3];
    const float* wout = (const float*)d_in[4];
    const float* bout = (const float*)d_in[5];
    float* out = (float*)d_out;

    const size_t per = (size_t)Bb * HEADS * Nn * HEAD;   // 8,388,608 f16 elements
    _Float16* qw   = (_Float16*)d_ws;                    // 16 MB
    _Float16* kw   = qw + per;                           // 16 MB
    _Float16* vw   = kw + per;                           // 16 MB (transposed [b,h,d,n])
    _Float16* o    = vw + per;                           // 16 MB
    _Float16* xnw  = o + per;                            //  2 MB  LN output f16
    _Float16* wq16 = xnw + (size_t)Bb * Nn * DIM;        // 0.75 MB
    _Float16* wo16 = wq16 + NQKV;                        // 0.25 MB -> ~67 MB total

    k_ln  <<<dim3(256),  dim3(256), 0, stream>>>(x, gam, bet, xnw);
    k_cvt <<<dim3(512),  dim3(256), 0, stream>>>(wqkv, wout, wq16, wo16);
    k_qkv <<<dim3(3072), dim3(256), 0, stream>>>(xnw, wq16, qw, kw, vw);
    k_attn<<<dim3(1024), dim3(128), 0, stream>>>(qw, kw, vw, o);
    k_proj<<<dim3(128),  dim3(256), 0, stream>>>(o, wo16, bout, out);
}
